// SequentialScorer_59820304499359
// MI455X (gfx1250) — compile-verified, matrix operand order checked by reading, not run on hardware
//
#include <hip/hip_runtime.h>
#include <hip/hip_bf16.h>

// Problem constants (from the reference)
#define BATCH   16384
#define MARK    1024
#define NFEAT   40
#define FEMB    16
#define KDIM    2688      // 2*1024 + 40*16
#define HIDDEN  4096

// Tiling
#define BLK_M   64        // batch rows per block
#define BLK_N   128       // hidden cols per block
#define KC      64        // K chunk staged in LDS
#define LDA     72        // padded LDS stride (ushorts), even -> 4B-aligned b32 loads

typedef __attribute__((ext_vector_type(16))) __bf16 v16bf;
typedef __attribute__((ext_vector_type(8)))  float  v8f;

__device__ __forceinline__ unsigned short f32_to_bf16(float x) {
    unsigned int u = __float_as_uint(x);
    u += 0x7FFFu + ((u >> 16) & 1u);        // round-to-nearest-even
    return (unsigned short)(u >> 16);
}

union FragU { unsigned int u[8]; v16bf v; };

// Load a 16x32 bf16 fragment (A layout, ISA 7.12.2) from LDS.
// For B (32x16 K-major) the same striping applies with "row" = N column.
// lanes 0-15 : row r0+lane,  K = kk+{0..7} and kk+{16..23}
// lanes 16-31: row r0+lane-16, K = kk+{8..15} and kk+{24..31}
__device__ __forceinline__ v16bf load_frag(const unsigned short* lds, int r0,
                                           int kk, int lane) {
    const int row = r0 + (lane & 15);
    const int kb  = kk + ((lane >> 4) << 3);        // +8 for upper half-wave
    const unsigned short* p = lds + row * LDA + kb;
    FragU f;
#pragma unroll
    for (int j = 0; j < 4; ++j) {
        f.u[j]     = *(const unsigned int*)(p + 2 * j);        // K = kb + 2j,2j+1
        f.u[j + 4] = *(const unsigned int*)(p + 2 * j + 16);   // K = kb+16 + 2j,..
    }
    return f.v;
}

__global__ void scorer_init_out(float* __restrict__ out,
                                const float* __restrict__ b2, int n) {
    int i = blockIdx.x * blockDim.x + threadIdx.x;
    if (i < n) out[i] = b2[0];
}

__global__ __launch_bounds__(256)
void scorer_fused_kernel(const float* __restrict__ emb_i,
                         const float* __restrict__ emb_a,
                         const int*   __restrict__ feat_mask,
                         const float* __restrict__ on_emb,
                         const float* __restrict__ off_emb,
                         const float* __restrict__ W1,
                         const float* __restrict__ bias1,
                         const float* __restrict__ W2,
                         float*       __restrict__ out) {
    __shared__ unsigned short Asm[BLK_M * LDA];   // bf16 activations (concat)
    __shared__ unsigned short Bsm[BLK_N * LDA];   // bf16 W1 tile

    const int tid    = threadIdx.x;
    const int n_base = blockIdx.x * BLK_N;
    const int m_base = blockIdx.y * BLK_M;

    const int wave = tid >> 5;
    const int lane = tid & 31;
    const int wm   = wave & 3;     // M sub-tile (16 rows)
    const int wn   = wave >> 2;    // N strip of 64 (0 or 1)

    v8f acc[4] = {};               // 4 x (16x16 f32) accumulators per wave

    const int ka    = tid & 63;    // this thread's fixed k within the chunk
    const int rbase = tid >> 6;    // 0..3

    for (int k0 = 0; k0 < KDIM; k0 += KC) {
        // ---- stage A tile: gather concat(emb_i, emb_a, feature-select) -> bf16
        {
            const int gk = k0 + ka;
#pragma unroll
            for (int s = 0; s < BLK_M / 4; ++s) {
                const int row = rbase + 4 * s;
                const int b   = m_base + row;
                float v;
                if (gk < MARK) {
                    v = emb_i[b * MARK + gk];
                } else if (gk < 2 * MARK) {
                    v = emb_a[b * MARK + (gk - MARK)];
                } else {
                    const int fk = gk - 2 * MARK;
                    const int f = fk >> 4, e = fk & 15;
                    v = (feat_mask[b * NFEAT + f] > 0) ? on_emb[f * FEMB + e]
                                                       : off_emb[f * FEMB + e];
                }
                Asm[row * LDA + ka] = f32_to_bf16(v);
            }
        }
        // ---- stage B tile: W1[n, k] -> bf16 (B is K x N = W1 transposed view)
        {
#pragma unroll
            for (int s = 0; s < BLK_N / 4; ++s) {
                const int n = rbase + 4 * s;
                const float* gp = W1 + (size_t)(n_base + n) * KDIM + (k0 + ka);
                Bsm[n * LDA + ka] = f32_to_bf16(*gp);
                if (k0 + KC < KDIM) __builtin_prefetch(gp + KC, 0, 1);
            }
        }
        __syncthreads();

#pragma unroll
        for (int kk = 0; kk < KC; kk += 32) {
            v16bf a = load_frag(Asm, wm * 16, kk, lane);
#pragma unroll
            for (int t = 0; t < 4; ++t) {
                v16bf b = load_frag(Bsm, wn * 64 + t * 16, kk, lane);
                acc[t] = __builtin_amdgcn_wmma_f32_16x16x32_bf16(
                    false, a, false, b, (short)0, acc[t], false, false);
            }
        }
        __syncthreads();
    }

    // ---- fused epilogue: out[b] += sum_n relu(h[b,n] + b1[n]) * W2[n]
    // D layout: lane<16 -> (M = wm*16 + v,   N = lane)
    //           lane>=16-> (M = wm*16 + 8+v, N = lane-16)
    float partial[8];
#pragma unroll
    for (int v = 0; v < 8; ++v) partial[v] = 0.f;

    const int nloc = lane & 15;
#pragma unroll
    for (int t = 0; t < 4; ++t) {
        const int col  = n_base + wn * 64 + t * 16 + nloc;
        const float bs = bias1[col];
        const float w2 = W2[col];
#pragma unroll
        for (int v = 0; v < 8; ++v)
            partial[v] += fmaxf(acc[t][v] + bs, 0.f) * w2;
    }
    // reduce the 16 N-lanes within each half-wave (xor 1,2,4,8 stays in-half)
#pragma unroll
    for (int m = 1; m <= 8; m <<= 1) {
#pragma unroll
        for (int v = 0; v < 8; ++v)
            partial[v] += __shfl_xor(partial[v], m, 32);
    }
    if (nloc == 0) {
        const int mrow = m_base + wm * 16 + (lane >> 4) * 8;
#pragma unroll
        for (int v = 0; v < 8; ++v)
            atomicAdd(&out[mrow + v], partial[v]);
    }
}

extern "C" void kernel_launch(void* const* d_in, const int* in_sizes, int n_in,
                              void* d_out, int out_size, void* d_ws, size_t ws_size,
                              hipStream_t stream) {
    const float* emb_i     = (const float*)d_in[0];
    const float* emb_a     = (const float*)d_in[1];
    const int*   feat_mask = (const int*)  d_in[2];
    const float* on_emb    = (const float*)d_in[3];
    const float* off_emb   = (const float*)d_in[4];
    const float* W1        = (const float*)d_in[5];
    const float* b1        = (const float*)d_in[6];
    const float* W2        = (const float*)d_in[7];
    const float* b2        = (const float*)d_in[8];
    float* out = (float*)d_out;

    (void)in_sizes; (void)n_in; (void)out_size; (void)d_ws; (void)ws_size;

    // out[b] = b2 (atomics accumulate on top; reset every call => replay-safe)
    scorer_init_out<<<(BATCH + 255) / 256, 256, 0, stream>>>(out, b2, BATCH);

    dim3 grid(HIDDEN / BLK_N, BATCH / BLK_M);   // 32 x 256 blocks
    scorer_fused_kernel<<<grid, 256, 0, stream>>>(
        emb_i, emb_a, feat_mask, on_emb, off_emb, W1, b1, W2, out);
}
